// GraphSAGE_78426102825061
// MI455X (gfx1250) — compile-verified, matrix operand order checked by reading, not run on hardware
//
#include <hip/hip_runtime.h>
#include <cstdint>
#include <cstddef>

typedef _Float16 half_t;
typedef _Float16 v16h __attribute__((ext_vector_type(16)));
typedef _Float16 v8h  __attribute__((ext_vector_type(8)));
typedef _Float16 v4h  __attribute__((ext_vector_type(4)));
typedef float    v8f  __attribute__((ext_vector_type(8)));

#define DIN   512
#define BATCH 1024
#define FAN1  25
#define FAN2  10

// ---------------------------------------------------------------------------
// CDNA5 async global->LDS copy (16B per lane), tracked by ASYNCcnt.
// lds_off: wave-relative LDS byte address (= low 32 bits of generic pointer).
// ---------------------------------------------------------------------------
__device__ __forceinline__ void async_ld16(uint32_t lds_off, const half_t* g) {
  asm volatile("global_load_async_to_lds_b128 %0, %1, off"
               :: "v"(lds_off), "v"((uint64_t)(uintptr_t)g)
               : "memory");
}

__device__ __forceinline__ void wait_async_then_barrier() {
#if __has_builtin(__builtin_amdgcn_s_wait_asynccnt)
  __builtin_amdgcn_s_wait_asynccnt(0);
#else
  asm volatile("s_wait_asynccnt 0" ::: "memory");
#endif
  __syncthreads();
}

// ---------------------------------------------------------------------------
// Weight prep: f32 [K][N] -> f16 [N][K]  (so WMMA B-fragments load contiguously)
// ---------------------------------------------------------------------------
__global__ void k_transpose_f16(const float* __restrict__ W, half_t* __restrict__ Wt,
                                int K, int N) {
  int i = blockIdx.x * blockDim.x + threadIdx.x;   // i = k*N + n -> coalesced read
  int k = i / N, n = i - k * N;
  Wt[(size_t)n * K + k] = (half_t)W[i];
}

// Wcat[n][k] (n<256, k<2048): k<1024 -> Ws1[k][n], else Wn1[k-1024][n]
__global__ void k_build_wcat(const float* __restrict__ Ws1, const float* __restrict__ Wn1,
                             half_t* __restrict__ Wcat) {
  int i = blockIdx.x * blockDim.x + threadIdx.x;   // i = k*256 + n
  int k = i >> 8, n = i & 255;
  float v = (k < 1024) ? Ws1[(size_t)k * 256 + n] : Wn1[(size_t)(k - 1024) * 256 + n];
  Wcat[(size_t)n * 2048 + k] = (half_t)v;
}

// ---------------------------------------------------------------------------
// Gather + mean over fanout: out[r][c] = scale * sum_f x[idx[r*fanout+f]][c]
// 128 threads/block, 4 f32 columns per thread (DIN=512). f16 output.
// This is the HBM-bound stage (~570 MB of gathered rows).
// ---------------------------------------------------------------------------
__global__ void k_gather_mean(const float* __restrict__ x, const int* __restrict__ idx,
                              int fanout, float scale, half_t* __restrict__ out) {
  const int r = blockIdx.x, t = threadIdx.x;
  const int* ip = idx + (size_t)r * fanout;
  float4 acc = make_float4(0.f, 0.f, 0.f, 0.f);
  for (int f = 0; f < fanout; ++f) {
    if (f + 1 < fanout)  // WGP-scope prefetch of next neighbor row
      __builtin_prefetch(x + (size_t)ip[f + 1] * DIN + t * 4, 0, 3);
    const float4* s = reinterpret_cast<const float4*>(x + (size_t)ip[f] * DIN) + t;
    float4 v = *s;
    acc.x += v.x; acc.y += v.y; acc.z += v.z; acc.w += v.w;
  }
  v4h o;
  o[0] = (half_t)(acc.x * scale); o[1] = (half_t)(acc.y * scale);
  o[2] = (half_t)(acc.z * scale); o[3] = (half_t)(acc.w * scale);
  *reinterpret_cast<v4h*>(out + (size_t)r * DIN + t * 4) = o;
}

// ---------------------------------------------------------------------------
// Mean over 25 contiguous rows of h1cat [25600][1024] -> hfull[r][1024 + c]
// ---------------------------------------------------------------------------
__global__ void k_mean25(const half_t* __restrict__ h1, half_t* __restrict__ hfull) {
  const int r = blockIdx.x, t = threadIdx.x;     // 256 threads, 4 cols each of 1024
  float a0 = 0.f, a1 = 0.f, a2 = 0.f, a3 = 0.f;
  for (int j = 0; j < FAN1; ++j) {
    v4h v = *reinterpret_cast<const v4h*>(h1 + ((size_t)r * FAN1 + j) * 1024 + t * 4);
    a0 += (float)v[0]; a1 += (float)v[1]; a2 += (float)v[2]; a3 += (float)v[3];
  }
  const float s = 1.f / (float)FAN1;
  v4h o;
  o[0] = (half_t)(a0 * s); o[1] = (half_t)(a1 * s);
  o[2] = (half_t)(a2 * s); o[3] = (half_t)(a3 * s);
  *reinterpret_cast<v4h*>(hfull + (size_t)r * 2048 + 1024 + t * 4) = o;
}

// ---------------------------------------------------------------------------
// WMMA f16 GEMM: C[M][N](+colofs) = act(A[M][K] * Bt[N][K]^T)
// Block tile 128x128, BK=32, 8 waves (4 in M x 2 in N), wave tile 32x64
// -> 8x v_wmma_f32_16x16x32_f16 per K-step. Double-buffered LDS filled by
// async global->LDS DMA; pipeline synced with s_wait_asynccnt + barrier.
// ---------------------------------------------------------------------------
template <bool RELU, typename OutT>
__global__ void __launch_bounds__(256)
k_wmma_gemm(const half_t* __restrict__ A, int lda,
            const half_t* __restrict__ Bt, int ldb,
            OutT* __restrict__ C, int ldc, int colofs, int Ktot) {
  constexpr int SA = 40;                       // padded LDS row stride (f16); 80B keeps b128 aligned
  __shared__ half_t As[2][128 * SA];
  __shared__ half_t Bs[2][128 * SA];

  const int tid  = threadIdx.x;
  const int wid  = tid >> 5;
  const int lane = tid & 31;
  const int wm   = wid & 3;                    // wave row group: 32 rows
  const int wn   = wid >> 2;                   // wave col group: 64 cols
  const int lr   = lane & 15;                  // M/N index within 16
  const int kh   = lane >> 4;                  // K-half select per ISA A/B layout

  const int rowBase = blockIdx.x * 128;
  const int colBase = blockIdx.y * 128;

  const int r   = tid >> 1;                    // staging row 0..127
  const int seg = (tid & 1) * 16;              // 0 / 16 f16 within the 32-wide K chunk

  const half_t* Ap = A  + (size_t)(rowBase + r) * lda + seg;
  const half_t* Bp = Bt + (size_t)(colBase + r) * ldb + seg;

  // wave-relative LDS byte addresses for this thread's staging slots
  const uint32_t aoff = (uint32_t)(uintptr_t)&As[0][r * SA + seg];
  const uint32_t boff = (uint32_t)(uintptr_t)&Bs[0][r * SA + seg];
  constexpr uint32_t BUFB = 128 * SA * sizeof(half_t);   // 10240 B per buffer

  v8f acc[2][4] = {};

  auto issue = [&](int k0, int buf) {
    const uint32_t ab = aoff + (uint32_t)buf * BUFB;
    const uint32_t bb = boff + (uint32_t)buf * BUFB;
    async_ld16(ab,      Ap + k0);
    async_ld16(ab + 16, Ap + k0 + 8);
    async_ld16(bb,      Bp + k0);
    async_ld16(bb + 16, Bp + k0 + 8);
  };

  issue(0, 0);
  wait_async_then_barrier();

  for (int k0 = 0; k0 < Ktot; k0 += 32) {
    const int cur = (k0 >> 5) & 1;
    if (k0 + 32 < Ktot) issue(k0 + 32, cur ^ 1);   // DMA next tile under compute

    const half_t* Ab = As[cur];
    const half_t* Bb = Bs[cur];

    // A fragment (16-bit 16x32 layout): lanes 0-15 K={0..7,16..23}, lanes 16-31 K={8..15,24..31}
    v16h af[2], bf[4];
#pragma unroll
    for (int mt = 0; mt < 2; ++mt) {
      const int row = wm * 32 + mt * 16 + lr;
      ((v8h*)&af[mt])[0] = *reinterpret_cast<const v8h*>(&Ab[row * SA + kh * 8]);
      ((v8h*)&af[mt])[1] = *reinterpret_cast<const v8h*>(&Ab[row * SA + kh * 8 + 16]);
    }
#pragma unroll
    for (int nt = 0; nt < 4; ++nt) {
      const int col = wn * 64 + nt * 16 + lr;
      ((v8h*)&bf[nt])[0] = *reinterpret_cast<const v8h*>(&Bb[col * SA + kh * 8]);
      ((v8h*)&bf[nt])[1] = *reinterpret_cast<const v8h*>(&Bb[col * SA + kh * 8 + 16]);
    }

#pragma unroll
    for (int mt = 0; mt < 2; ++mt)
#pragma unroll
      for (int nt = 0; nt < 4; ++nt)
        acc[mt][nt] = __builtin_amdgcn_wmma_f32_16x16x32_f16(
            false, af[mt], false, bf[nt], (short)0, acc[mt][nt], false, false);

    wait_async_then_barrier();
  }

  // Epilogue per C/D layout: VGPR v -> M = kh*8 + v, lane%16 -> N
#pragma unroll
  for (int mt = 0; mt < 2; ++mt) {
    const int row0 = rowBase + wm * 32 + mt * 16 + kh * 8;
#pragma unroll
    for (int nt = 0; nt < 4; ++nt) {
      const int col = colofs + colBase + wn * 64 + nt * 16 + lr;
#pragma unroll
      for (int v = 0; v < 8; ++v) {
        float val = acc[mt][nt][v];
        if (RELU) val = val > 0.f ? val : 0.f;
        C[(size_t)(row0 + v) * ldc + col] = (OutT)val;
      }
    }
  }
}

// ---------------------------------------------------------------------------
extern "C" void kernel_launch(void* const* d_in, const int* in_sizes, int n_in,
                              void* d_out, int out_size, void* d_ws, size_t ws_size,
                              hipStream_t stream) {
  const float* x    = (const float*)d_in[0];
  const int* nodes  = (const int*)d_in[1];
  const int* neigh1 = (const int*)d_in[2];
  const int* neigh2 = (const int*)d_in[3];
  const float* Ws0  = (const float*)d_in[4];
  const float* Wn0  = (const float*)d_in[5];
  const float* Ws1  = (const float*)d_in[6];
  const float* Wn1  = (const float*)d_in[7];
  float* out = (float*)d_out;
  (void)in_sizes; (void)n_in; (void)out_size; (void)ws_size;

  char* p = (char*)d_ws;
  auto alloc = [&](size_t bytes) -> char* {
    char* q = p; p += (bytes + 255) & ~(size_t)255; return q;
  };
  half_t* ws0t  = (half_t*)alloc((size_t)512 * 512 * 2);
  half_t* wn0t  = (half_t*)alloc((size_t)512 * 512 * 2);
  half_t* wcat  = (half_t*)alloc((size_t)2048 * 256 * 2);
  half_t* xg1   = (half_t*)alloc((size_t)25600 * 512 * 2);   // x[neigh1] f16
  half_t* mean2 = (half_t*)alloc((size_t)25600 * 512 * 2);   // mean_10 x[neigh2]
  half_t* xg0   = (half_t*)alloc((size_t)1024 * 512 * 2);    // x[nodes]
  half_t* mean1 = (half_t*)alloc((size_t)1024 * 512 * 2);    // mean_25 x[neigh1]
  half_t* h1cat = (half_t*)alloc((size_t)25600 * 1024 * 2);  // layer0 1-hop output
  half_t* hfull = (half_t*)alloc((size_t)1024 * 2048 * 2);   // [h0cat | mean25(h1cat)]

  // weights
  k_transpose_f16<<<(512 * 512) / 256, 256, 0, stream>>>(Ws0, ws0t, 512, 512);
  k_transpose_f16<<<(512 * 512) / 256, 256, 0, stream>>>(Wn0, wn0t, 512, 512);
  k_build_wcat<<<(2048 * 256) / 256, 256, 0, stream>>>(Ws1, Wn1, wcat);

  // gathers / neighbor means (memory-bound stage)
  k_gather_mean<<<25600, 128, 0, stream>>>(x, neigh1, 1, 1.f, xg1);
  k_gather_mean<<<25600, 128, 0, stream>>>(x, neigh2, FAN2, 1.f / FAN2, mean2);
  k_gather_mean<<<1024, 128, 0, stream>>>(x, nodes, 1, 1.f, xg0);
  k_gather_mean<<<1024, 128, 0, stream>>>(x, neigh1, FAN1, 1.f / FAN1, mean1);

  // layer 0 (concat=True, relu): 1-hop nodes, M=25600, K=512, N=512 each half
  k_wmma_gemm<true, half_t><<<dim3(200, 4), 256, 0, stream>>>(xg1,   512, ws0t, 512, h1cat, 1024,   0, 512);
  k_wmma_gemm<true, half_t><<<dim3(200, 4), 256, 0, stream>>>(mean2, 512, wn0t, 512, h1cat, 1024, 512, 512);
  // layer 0: roots, M=1024 -> hfull cols [0,1024)
  k_wmma_gemm<true, half_t><<<dim3(8, 4), 256, 0, stream>>>(xg0,   512, ws0t, 512, hfull, 2048,   0, 512);
  k_wmma_gemm<true, half_t><<<dim3(8, 4), 256, 0, stream>>>(mean1, 512, wn0t, 512, hfull, 2048, 512, 512);

  // mean over fanout 25 of h1cat -> hfull cols [1024,2048)
  k_mean25<<<1024, 256, 0, stream>>>(h1cat, hfull);

  // layer 1 (concat=False => sum) as one GEMM: out = hfull @ [Ws1;Wn1], M=1024,K=2048,N=256, f32 out
  k_wmma_gemm<false, float><<<dim3(8, 2), 256, 0, stream>>>(hfull, 2048, wcat, 2048, out, 256, 0, 2048);
}
